// MultiheadAttention_13030930776120
// MI455X (gfx1250) — matrix operand order repaired and checked by reading, not compiled, not run on hardware
//
#include <hip/hip_runtime.h>
#include <stdint.h>

#define NH    16
#define NB    2
#define SEQ   2048
#define DM    1024
#define DK    64

typedef __attribute__((ext_vector_type(16))) __bf16        v16bf;
typedef __attribute__((ext_vector_type(8)))  float         v8f;
typedef __attribute__((ext_vector_type(8)))  unsigned int  v8u;
typedef __attribute__((ext_vector_type(4)))  unsigned int  v4u;
typedef __attribute__((ext_vector_type(4)))  unsigned int  u32x4;
typedef __attribute__((ext_vector_type(8)))  int           i32x8;
typedef __attribute__((ext_vector_type(4)))  int           i32x4;

// ---------- helpers ----------------------------------------------------------

__device__ __forceinline__ v8f vzero8() {
  v8f z;
  #pragma unroll
  for (int i = 0; i < 8; ++i) z[i] = 0.f;
  return z;
}
__device__ __forceinline__ v8f wmma_bf16(v16bf a, v16bf b, v8f c) {
  // (neg_a, A, neg_b, B, c_mod, C, reuse_a, reuse_b)
  return __builtin_amdgcn_wmma_f32_16x16x32_bf16(false, a, false, b, (short)0, c,
                                                 false, false);
}

// A fragment (16x32 bf16). Lane row fixed by caller; K pattern per ISA:
// lanes 0-15: K in {kb..kb+7} (V0-3) and {kb+16..kb+23} (V4-7); lanes 16-31: +8.
__device__ __forceinline__ v16bf a_from_bf16(const __bf16* row, int kb, int lane) {
  int off = kb + ((lane & 16) ? 8 : 0);
  v4u lo = *(const v4u*)(row + off);
  v4u hi = *(const v4u*)(row + off + 16);
  v8u r; r[0]=lo[0]; r[1]=lo[1]; r[2]=lo[2]; r[3]=lo[3];
         r[4]=hi[0]; r[5]=hi[1]; r[6]=hi[2]; r[7]=hi[3];
  return __builtin_bit_cast(v16bf, r);
}
__device__ __forceinline__ v16bf a_from_f32(const float* row, int kb, int lane) {
  int off = kb + ((lane & 16) ? 8 : 0);
  const float4* p0 = (const float4*)(row + off);
  const float4* p1 = (const float4*)(row + off + 16);
  float4 x0 = p0[0], x1 = p0[1], y0 = p1[0], y1 = p1[1];
  v16bf r;                       // native v_cvt_pk_bf16_f32 path
  r[0]=(__bf16)x0.x; r[1]=(__bf16)x0.y; r[2]=(__bf16)x0.z; r[3]=(__bf16)x0.w;
  r[4]=(__bf16)x1.x; r[5]=(__bf16)x1.y; r[6]=(__bf16)x1.z; r[7]=(__bf16)x1.w;
  r[8]=(__bf16)y0.x; r[9]=(__bf16)y0.y; r[10]=(__bf16)y0.z; r[11]=(__bf16)y0.w;
  r[12]=(__bf16)y1.x; r[13]=(__bf16)y1.y; r[14]=(__bf16)y1.z; r[15]=(__bf16)y1.w;
  return r;
}
// B fragment (32x16 bf16). Lane = column n; K: lanes0-15 {kb..kb+7,kb+16..kb+23},
// lanes16-31: +8. "row" = consecutive K elements for lane's column.
__device__ __forceinline__ v16bf b_from_bf16(const __bf16* row, int kb, int lane) {
  int off = kb + ((lane & 16) ? 8 : 0);
  v4u lo = *(const v4u*)(row + off);
  v4u hi = *(const v4u*)(row + off + 16);
  v8u r; r[0]=lo[0]; r[1]=lo[1]; r[2]=lo[2]; r[3]=lo[3];
         r[4]=hi[0]; r[5]=hi[1]; r[6]=hi[2]; r[7]=hi[3];
  return __builtin_bit_cast(v16bf, r);
}

// ---------- kernel 0: weight transpose + fp32->bf16 convert ------------------
// wqt/wkt/wvt: [h][n][d] (k-contiguous), pwt: [n][k] (already k-contiguous).
// grid: (4096, 4) block: 256; one element per thread, coalesced writes.

__global__ __launch_bounds__(256) void prep_weights_kernel(
    const float* __restrict__ wq, const float* __restrict__ wk,
    const float* __restrict__ wv, const float* __restrict__ pw,
    __bf16* __restrict__ wqt, __bf16* __restrict__ wkt,
    __bf16* __restrict__ wvt, __bf16* __restrict__ pwt)
{
  const int sel = blockIdx.y;
  const unsigned i = blockIdx.x * 256 + threadIdx.x;       // < 1,048,576
  if (sel == 3) {
    pwt[i] = (__bf16)pw[i];                                // plain convert
  } else {
    const float* W = (sel == 0) ? wq : (sel == 1) ? wk : wv;
    __bf16* WT = (sel == 0) ? wqt : (sel == 1) ? wkt : wvt;
    const unsigned h = i >> 16, r = i & 65535;
    const unsigned n = r >> 10, d = r & 1023;              // WT[h][n][d] = W[h][d][n]
    WT[i] = (__bf16)W[(size_t)h * (DM * DK) + (size_t)d * DK + n];
  }
}

// ---------- kernel 1: per-head Q/K/V projections (bf16 WMMA GEMM) ------------
// grid: (M/128, NH, 3)  block: 256 (8 waves). Wave computes a 16x64 tile.
// V is stored TRANSPOSED ([h][b][dv][s]) so the PV B-fragments are contiguous.

__global__ __launch_bounds__(256) void qkv_proj_kernel(
    const float* __restrict__ q, const float* __restrict__ k, const float* __restrict__ v,
    const __bf16* __restrict__ wqt, const __bf16* __restrict__ wkt,
    const __bf16* __restrict__ wvt,
    __bf16* __restrict__ qs, __bf16* __restrict__ ks, __bf16* __restrict__ vt)
{
  const int lane = threadIdx.x & 31, wave = threadIdx.x >> 5;
  const int h = blockIdx.y, sel = blockIdx.z;
  const float* X = (sel == 0) ? q : (sel == 1) ? k : v;
  const __bf16* WT =
      ((sel == 0) ? wqt : (sel == 1) ? wkt : wvt) + (size_t)h * DK * DM;  // [n][d]

  const int m0 = blockIdx.x * 128 + wave * 16;
  const float* xrow = X + (size_t)(m0 + (lane & 15)) * DM;

  v8f acc[4];
  #pragma unroll
  for (int t = 0; t < 4; ++t) acc[t] = vzero8();

  for (int kb = 0; kb < DM; kb += 32) {
    __builtin_prefetch(xrow + kb + 32, 0, 0);              // global_prefetch_b8
    v16bf a = a_from_f32(xrow, kb, lane);
    #pragma unroll
    for (int nt = 0; nt < 4; ++nt) {
      const __bf16* wrow = WT + (size_t)(nt * 16 + (lane & 15)) * DM;
      v16bf b = b_from_bf16(wrow, kb, lane);
      acc[nt] = wmma_bf16(a, b, acc[nt]);
    }
  }

  const int bi = m0 >> 11, sbase = m0 & 2047;
  const int rsh = (lane & 16) ? 8 : 0;
  if (sel < 2) {
    // head-major bf16: Y[((h*NB+bi)*SEQ + s)*DK + col]
    __bf16* Y = ((sel == 0) ? qs : ks) + ((size_t)(h * NB + bi) * SEQ + sbase) * DK;
    #pragma unroll
    for (int nt = 0; nt < 4; ++nt) {
      int col = nt * 16 + (lane & 15);
      #pragma unroll
      for (int j = 0; j < 8; ++j)
        Y[(size_t)(j + rsh) * DK + col] = (__bf16)acc[nt][j];
    }
  } else {
    // transposed: vt[((h*NB+bi)*DK + col)*SEQ + s]
    __bf16* Yt = vt + (size_t)(h * NB + bi) * DK * SEQ;
    #pragma unroll
    for (int nt = 0; nt < 4; ++nt) {
      int col = nt * 16 + (lane & 15);
      __bf16* p = Yt + (size_t)col * SEQ + sbase;
      #pragma unroll
      for (int j = 0; j < 8; ++j)
        p[j + rsh] = (__bf16)acc[nt][j];
    }
  }
}

// ---------- kernel 2: fused scores + softmax + attns write + PV --------------
// grid: (SEQ/16, NB, NH)  block: 256 (8 waves). Each wave owns a 16x256 key chunk.
// The 16x64 Q tile is DMA'd into LDS once per workgroup via the Tensor Data Mover.

__global__ __launch_bounds__(256) void attn_kernel(
    const __bf16* __restrict__ qs, const __bf16* __restrict__ ks,
    const __bf16* __restrict__ vt, float* __restrict__ attns,
    __bf16* __restrict__ ob)
{
  __shared__ __bf16 qtile[16 * DK];              // 2 KB, row-major 16x64
  __shared__ float redmax[8][16];
  __shared__ float redsum[8][16];
  __shared__ float Oacc[16 * DK];

  const int lane = threadIdx.x & 31, wave = threadIdx.x >> 5;
  const int qt = blockIdx.x, bi = blockIdx.y, h = blockIdx.z;
  const size_t hb = (size_t)(h * NB + bi);
  const __bf16* Q  = qs + hb * SEQ * DK;
  const __bf16* K  = ks + hb * SEQ * DK;
  const __bf16* Vt = vt + hb * DK * SEQ;        // [dv][s]
  float* A = attns + hb * SEQ * SEQ;            // attns[hb][q][s]

  for (int i = threadIdx.x; i < 16 * DK; i += 256) Oacc[i] = 0.f;

  const int q0 = qt * 16;
  const int rsh = (lane & 16) ? 8 : 0;

  // --- TDM: copy contiguous 1024-element (2 KB) Q tile into LDS (one wave) ---
  if (wave == 0) {
    const unsigned long long ga =
        (unsigned long long)(uintptr_t)(const void*)(Q + (size_t)q0 * DK);
    const unsigned ldsoff = (unsigned)(uintptr_t)(void*)&qtile[0];  // LDS byte offset
    u32x4 g0;                                    // D# group 0 (ISA 8.3)
    g0[0] = 1u;                                  // count=1, user descriptor
    g0[1] = ldsoff;                              // lds_addr
    g0[2] = (unsigned)(ga & 0xffffffffu);        // global_addr[31:0]
    g0[3] = (unsigned)(ga >> 32) | (2u << 30);   // global_addr[56:32] | type=2
    i32x8 g1;                                    // D# group 1 (ISA 8.4)
    g1[0] = (int)(1u << 16);                     // wg_mask=0, data_size=1 (2 bytes)
    g1[1] = (int)(1024u << 16);                  // tensor_dim0[15:0]=1024
    g1[2] = (int)(1u << 16);                     // tensor_dim0[31:16]=0, tensor_dim1=1
    g1[3] = (int)(1024u << 16);                  // tensor_dim1 hi=0, tile_dim0=1024
    g1[4] = 1;                                   // tile_dim1=1, tile_dim2=0
    g1[5] = 1024;                                // tensor_dim0_stride
    g1[6] = 0; g1[7] = 0;                        // dim1_stride=0
    i32x4 g2 = {0, 0, 0, 0};
    i32x4 g3 = {0, 0, 0, 0};
#if defined(__clang_major__) && __clang_major__ >= 23
    i32x8 g4 = {0, 0, 0, 0, 0, 0, 0, 0};
    __builtin_amdgcn_tensor_load_to_lds(g0, g1, g2, g3, g4, 0);
#else
    __builtin_amdgcn_tensor_load_to_lds(g0, g1, g2, g3, 0);
#endif
    __builtin_amdgcn_s_wait_tensorcnt(0);
  }
  __syncthreads();

  const __bf16* qrow = qtile + (lane & 15) * DK;
  v16bf aq[2];
  aq[0] = a_from_bf16(qrow, 0, lane);
  aq[1] = a_from_bf16(qrow, 32, lane);

  const int key0 = wave * 256;
  v8f acc[16];
  #pragma unroll
  for (int t = 0; t < 16; ++t) acc[t] = vzero8();
  // k-half outer / tile inner: consecutive WMMAs hit independent accumulators
  #pragma unroll
  for (int half = 0; half < 2; ++half) {
    #pragma unroll
    for (int t = 0; t < 16; ++t) {
      const __bf16* krow = K + (size_t)(key0 + t * 16 + (lane & 15)) * DK;
      v16bf b = b_from_bf16(krow, half * 32, lane);
      acc[t] = wmma_bf16(aq[half], b, acc[t]);
    }
  }
  const float scale = 0.03125f;                 // 1/sqrt(D_MODEL) = 1/32
  #pragma unroll
  for (int t = 0; t < 16; ++t)
    #pragma unroll
    for (int j = 0; j < 8; ++j) acc[t][j] *= scale;

  // ---- softmax: lane-group shuffle reduce, then cross-wave via LDS ----
  float m[8];
  #pragma unroll
  for (int j = 0; j < 8; ++j) {
    float mv = -3.4e38f;
    #pragma unroll
    for (int t = 0; t < 16; ++t) mv = fmaxf(mv, acc[t][j]);
    mv = fmaxf(mv, __shfl_xor(mv, 1));
    mv = fmaxf(mv, __shfl_xor(mv, 2));
    mv = fmaxf(mv, __shfl_xor(mv, 4));
    mv = fmaxf(mv, __shfl_xor(mv, 8));
    m[j] = mv;
  }
  if ((lane & 15) == 0) {
    #pragma unroll
    for (int j = 0; j < 8; ++j) redmax[wave][rsh + j] = m[j];
  }
  __syncthreads();
  float rowm[8];
  #pragma unroll
  for (int j = 0; j < 8; ++j) {
    float mv = -3.4e38f;
    #pragma unroll
    for (int w = 0; w < 8; ++w) mv = fmaxf(mv, redmax[w][rsh + j]);
    rowm[j] = mv;
  }
  float s[8];
  #pragma unroll
  for (int j = 0; j < 8; ++j) s[j] = 0.f;
  #pragma unroll
  for (int t = 0; t < 16; ++t)
    #pragma unroll
    for (int j = 0; j < 8; ++j) {
      float e = __expf(acc[t][j] - rowm[j]);
      acc[t][j] = e;
      s[j] += e;
    }
  #pragma unroll
  for (int j = 0; j < 8; ++j) {
    float sv = s[j];
    sv += __shfl_xor(sv, 1);
    sv += __shfl_xor(sv, 2);
    sv += __shfl_xor(sv, 4);
    sv += __shfl_xor(sv, 8);
    s[j] = sv;
  }
  if ((lane & 15) == 0) {
    #pragma unroll
    for (int j = 0; j < 8; ++j) redsum[wave][rsh + j] = s[j];
  }
  __syncthreads();
  float inv[8];
  #pragma unroll
  for (int j = 0; j < 8; ++j) {
    float sv = 0.f;
    #pragma unroll
    for (int w = 0; w < 8; ++w) sv += redsum[w][rsh + j];
    inv[j] = 1.0f / sv;
  }

  // ---- write normalized probabilities (attns output), this wave's chunk ----
  #pragma unroll
  for (int t = 0; t < 16; ++t) {
    int col = key0 + t * 16 + (lane & 15);
    #pragma unroll
    for (int j = 0; j < 8; ++j) {
      float p = acc[t][j] * inv[j];
      A[(size_t)(q0 + j + rsh) * SEQ + col] = p;
    }
  }

  // Same-wave store->load ordering (ISA 7.3): drain stores before reading P back.
  asm volatile("s_wait_storecnt 0x0" ::: "memory");

  // ---- PV: O(16x64) partial over this wave's 256 keys; V^T gives contiguous B ----
  v8f o[4];
  #pragma unroll
  for (int nt = 0; nt < 4; ++nt) o[nt] = vzero8();
  const float* prow = A + (size_t)(q0 + (lane & 15)) * SEQ;
  #pragma unroll
  for (int kb = 0; kb < 8; ++kb) {
    v16bf pa = a_from_f32(prow, key0 + kb * 32, lane);
    #pragma unroll
    for (int nt = 0; nt < 4; ++nt) {
      const __bf16* vrow = Vt + (size_t)(nt * 16 + (lane & 15)) * SEQ;
      v16bf vb = b_from_bf16(vrow, key0 + kb * 32, lane);
      o[nt] = wmma_bf16(pa, vb, o[nt]);
    }
  }
  #pragma unroll
  for (int nt = 0; nt < 4; ++nt) {
    int col = nt * 16 + (lane & 15);
    #pragma unroll
    for (int j = 0; j < 8; ++j)
      atomicAdd(&Oacc[(j + rsh) * DK + col], o[nt][j]);   // ds_add_f32
  }
  __syncthreads();

  // concat-head layout (B, S, NH*DK) bf16 for the output projection
  for (int i = threadIdx.x; i < 16 * DK; i += 256) {
    int r = i >> 6, c = i & 63;
    ob[(size_t)(bi * SEQ + q0 + r) * (NH * DK) + h * DK + c] = (__bf16)Oacc[i];
  }
}

// ---------- kernel 3: output projection  out = O @ proj_w^T + b --------------
// grid: (M/128, N/64)  block: 256 (8 waves). B-fragments contiguous from pwt.

__global__ __launch_bounds__(256) void out_proj_kernel(
    const __bf16* __restrict__ ob, const __bf16* __restrict__ pwt,
    const float* __restrict__ pb, float* __restrict__ out)
{
  const int lane = threadIdx.x & 31, wave = threadIdx.x >> 5;
  const int m0 = blockIdx.x * 128 + wave * 16;
  const int n0 = blockIdx.y * 64;
  const __bf16* arow = ob + (size_t)(m0 + (lane & 15)) * DM;

  v8f acc[4];
  #pragma unroll
  for (int t = 0; t < 4; ++t) acc[t] = vzero8();

  for (int kb = 0; kb < DM; kb += 32) {
    __builtin_prefetch(arow + kb + 32, 0, 0);    // global_prefetch_b8
    v16bf a = a_from_bf16(arow, kb, lane);
    #pragma unroll
    for (int nt = 0; nt < 4; ++nt) {
      // out = O @ W^T => B[k][n] = W[n][k]: row n of pwt, contiguous k
      const __bf16* wrow = pwt + (size_t)(n0 + nt * 16 + (lane & 15)) * DM;
      v16bf b = b_from_bf16(wrow, kb, lane);
      acc[nt] = wmma_bf16(a, b, acc[nt]);
    }
  }
  const int rsh = (lane & 16) ? 8 : 0;
  #pragma unroll
  for (int nt = 0; nt < 4; ++nt) {
    int n = n0 + nt * 16 + (lane & 15);
    float bias = pb[n];
    #pragma unroll
    for (int j = 0; j < 8; ++j)
      out[(size_t)(m0 + j + rsh) * DM + n] = acc[nt][j] + bias;
  }
}

// ---------- launch -----------------------------------------------------------

extern "C" void kernel_launch(void* const* d_in, const int* in_sizes, int n_in,
                              void* d_out, int out_size, void* d_ws, size_t ws_size,
                              hipStream_t stream) {
  const float* q  = (const float*)d_in[0];
  const float* k  = (const float*)d_in[1];
  const float* v  = (const float*)d_in[2];
  const float* wq = (const float*)d_in[3];
  const float* wk = (const float*)d_in[4];
  const float* wv = (const float*)d_in[5];
  const float* pw = (const float*)d_in[6];
  const float* pb = (const float*)d_in[7];

  float* out   = (float*)d_out;                       // (B,S,DM) = 4,194,304 floats
  float* attns = out + (size_t)NB * SEQ * DM;         // (NH*NB,S,S)

  const size_t headElems = (size_t)NH * NB * SEQ * DK;  // 4,194,304 bf16 each
  const size_t wElems    = (size_t)NH * DK * DM;        // 1,048,576 bf16 each
  __bf16* qs  = (__bf16*)d_ws;
  __bf16* ks  = qs  + headElems;
  __bf16* vt  = ks  + headElems;                        // V transposed [h][b][dv][s]
  __bf16* ob  = vt  + headElems;                        // (B*S, NH*DK)
  __bf16* wqt = ob  + headElems;                        // [h][n][d]
  __bf16* wkt = wqt + wElems;
  __bf16* wvt = wkt + wElems;
  __bf16* pwt = wvt + wElems;                           // [n][k]

  prep_weights_kernel<<<dim3(4096, 4), 256, 0, stream>>>(
      wq, wk, wv, pw, wqt, wkt, wvt, pwt);
  qkv_proj_kernel<<<dim3((NB * SEQ) / 128, NH, 3), 256, 0, stream>>>(
      q, k, v, wqt, wkt, wvt, qs, ks, vt);
  attn_kernel<<<dim3(SEQ / 16, NB, NH), 256, 0, stream>>>(qs, ks, vt, attns, ob);
  out_proj_kernel<<<dim3((NB * SEQ) / 128, DM / 64), 256, 0, stream>>>(
      ob, pwt, pb, out);
}
